// ContextualAttention_36945308680514
// MI455X (gfx1250) — matrix operand order repaired and checked by reading, not compiled, not run on hardware
//
#include <hip/hip_runtime.h>
#include <stdint.h>

// ---------------------------------------------------------------------------
// Contextual attention (DeepFill) on MI455X / gfx1250.
// Two big GEMMs done with v_wmma_f32_16x16x32_bf16 (f32 accumulate).
// ---------------------------------------------------------------------------

typedef __attribute__((ext_vector_type(16))) __bf16    v16bf;
typedef __attribute__((ext_vector_type(8)))  float     v8f;
typedef __attribute__((ext_vector_type(8)))  uint32_t  v8u;

static constexpr int Cc   = 128;      // channels
static constexpr int Hh   = 128, Ww = 128;
static constexpr int HS   = 64,  WS = 64;   // downsampled
static constexpr int Lp   = 4096;           // number of patches / positions
static constexpr int K1D  = Cc * 9;         // 1152  (3x3 patch features)
static constexpr int K2D  = Cc * 16;        // 2048  (4x4 raw patch features)
static constexpr float SCALE_F = 10.0f;

__device__ __forceinline__ uint16_t f2bf(float f) {
    uint32_t u = __builtin_bit_cast(uint32_t, f);
    u += 0x7FFFu + ((u >> 16) & 1u);          // round-to-nearest-even
    return (uint16_t)(u >> 16);
}

// ---------------------------------------------------------------------------
// Stage 1: 3x3 patch matrix P[4096][1152] (bf16), row scale rs, mask mm.
// fb[c,y,x] = x[b,c,2y,2x]; SAME padding (pad 1).
// ---------------------------------------------------------------------------
__global__ void k_prep3(const float* __restrict__ x, const float* __restrict__ mask,
                        uint16_t* __restrict__ P, float* __restrict__ rs,
                        float* __restrict__ mmv, int b)
{
    int p  = blockIdx.x;          // patch id 0..4095
    int py = p >> 6, px = p & 63;
    int c  = threadIdx.x;         // 128 threads = channel
    const float* xb = x + ((size_t)b * Cc + c) * (size_t)(Hh * Ww);

    float vals[9];
    float ss = 0.f;
#pragma unroll
    for (int dy = 0; dy < 3; ++dy)
#pragma unroll
        for (int dx = 0; dx < 3; ++dx) {
            int yy = py + dy - 1, xx = px + dx - 1;
            float v = 0.f;
            if (yy >= 0 && yy < HS && xx >= 0 && xx < WS)
                v = xb[(size_t)(2 * yy) * Ww + 2 * xx];
            vals[dy * 3 + dx] = v;
            ss += v * v;
        }
    uint16_t* prow = P + (size_t)p * K1D + c * 9;
#pragma unroll
    for (int t = 0; t < 9; ++t) prow[t] = f2bf(vals[t]);

    __shared__ float red[128];
    red[c] = ss;
    __syncthreads();
    for (int s = 64; s > 0; s >>= 1) {
        if (c < s) red[c] += red[c + s];
        __syncthreads();
    }
    if (c == 0) {
        float nrm = sqrtf(red[0]);
        const float* mb = mask + (size_t)b * (size_t)(Hh * Ww);
        float msum = 0.f;
        for (int dy = 0; dy < 3; ++dy)
            for (int dx = 0; dx < 3; ++dx) {
                int yy = py + dy - 1, xx = px + dx - 1;
                if (yy >= 0 && yy < HS && xx >= 0 && xx < WS)
                    msum += mb[(size_t)(2 * yy) * Ww + 2 * xx];
            }
        float mm = (msum == 0.f) ? 1.f : 0.f;
        mmv[p] = mm;
        rs[p]  = mm * SCALE_F / fmaxf(nrm, 1e-4f);
    }
}

// ---------------------------------------------------------------------------
// Stage 2: raw 4x4 stride-2 patches W4[4096][2048] (bf16), SAME pad (pad 1).
// ---------------------------------------------------------------------------
__global__ void k_prep4(const float* __restrict__ x, uint16_t* __restrict__ W4, int b)
{
    int p  = blockIdx.x;
    int py = p >> 6, px = p & 63;
    int c  = threadIdx.x;         // 128
    const float* xb = x + ((size_t)b * Cc + c) * (size_t)(Hh * Ww);
    uint16_t* wrow = W4 + (size_t)p * K2D + c * 16;
#pragma unroll
    for (int a = 0; a < 4; ++a)
#pragma unroll
        for (int bb = 0; bb < 4; ++bb) {
            int yy = 2 * py + a - 1, xx = 2 * px + bb - 1;
            float v = (yy >= 0 && yy < Hh && xx >= 0 && xx < Ww)
                          ? xb[(size_t)yy * Ww + xx] : 0.f;
            wrow[a * 4 + bb] = f2bf(v);
        }
}

// ---------------------------------------------------------------------------
// Tiled bf16 GEMM, f32 accumulate. Cm[M x N] = A[M x K] * B[K x N].
// If TRANSB, element B(k,n) = Bsrc[n*ldb + k]; else Bsrc[k*ldb + n].
// Block tile 64(M) x 128(N) x 32(K); 256 threads = 8 wave32 waves (2x4),
// each wave computes a 32x32 subtile via 2x2 v_wmma_f32_16x16x32_bf16.
// All dims are multiples of the tile -> no predication, EXEC all-ones.
// ---------------------------------------------------------------------------
template <bool TRANSB>
__global__ void __launch_bounds__(256)
k_gemm_bf16(const uint16_t* __restrict__ A, int lda,
            const uint16_t* __restrict__ Bsrc, int ldb,
            float* __restrict__ Cm, int ldc, int Kdim)
{
    __shared__ uint32_t As[64][16];   // [m][k dword pair]  : 4 KB
    __shared__ uint16_t Bs[128][32];  // [n][k]             : 8 KB

    const int t    = threadIdx.x;
    const int lane = t & 31;
    const int wave = t >> 5;     // 0..7
    const int wm   = wave >> 2;  // 0..1 (M)
    const int wn   = wave & 3;   // 0..3 (N)
    const int half = lane >> 4;  // 0/1
    const int l15  = lane & 15;

    const int m0 = blockIdx.y * 64;
    const int n0 = blockIdx.x * 128;

    const int ar = t >> 2;        // A load row 0..63
    const int ak = (t & 3) * 8;   // A load k offset (8 bf16 = 16 B)

    v8f acc[2][2] = {};

    for (int k0 = 0; k0 < Kdim; k0 += 32) {
        // ---- stage A tile [64 x 32] ----
        {
            const uint16_t* src = A + (size_t)(m0 + ar) * lda + (k0 + ak);
            uint4 v = *reinterpret_cast<const uint4*>(src);
            *reinterpret_cast<uint4*>(&As[ar][(t & 3) * 4]) = v;
            if (k0 + 32 < Kdim)
                __builtin_prefetch(src + 32, 0, 3);   // global_prefetch_b8
        }
        // ---- stage B tile into Bs[n][k] ----
        if (TRANSB) {
            int nr = t >> 1;            // 0..127
            int kk = (t & 1) * 16;      // 16 bf16 = 32 B
            const uint16_t* src = Bsrc + (size_t)(n0 + nr) * ldb + (k0 + kk);
            const uint4* s4 = reinterpret_cast<const uint4*>(src);
            uint4 v0 = s4[0], v1 = s4[1];
            uint4* d4 = reinterpret_cast<uint4*>(&Bs[nr][kk]);
            d4[0] = v0; d4[1] = v1;
            if (k0 + 32 < Kdim)
                __builtin_prefetch(src + 32, 0, 3);
        } else {
            int kr = t >> 3;            // 0..31
            int nn = (t & 7) * 16;      // 16 bf16 along N
            const uint16_t* src = Bsrc + (size_t)(k0 + kr) * ldb + (n0 + nn);
            alignas(16) uint16_t tmp[16];
            const uint4* s4 = reinterpret_cast<const uint4*>(src);
            uint4 v0 = s4[0], v1 = s4[1];
            *reinterpret_cast<uint4*>(tmp)     = v0;
            *reinterpret_cast<uint4*>(tmp + 8) = v1;
#pragma unroll
            for (int i = 0; i < 16; ++i) Bs[nn + i][kr] = tmp[i];
            if (k0 + 32 < Kdim)
                __builtin_prefetch(Bsrc + (size_t)(k0 + 32 + kr) * ldb + (n0 + nn), 0, 3);
        }
        __syncthreads();

        // ---- fragments per CDNA5 wave32 layouts ----
        v16bf afrag[2], bfrag[2];
#pragma unroll
        for (int i = 0; i < 2; ++i) {
            // A 16x32: lane m = l15; dwords 0..3 -> K = half*8 + 2v,
            //          dwords 4..7 -> K = 16 + half*8 + 2v
            int m = wm * 32 + i * 16 + l15;
            v8u u;
#pragma unroll
            for (int v = 0; v < 4; ++v) {
                u[v]     = As[m][half * 4 + v];
                u[4 + v] = As[m][8 + half * 4 + v];
            }
            afrag[i] = __builtin_bit_cast(v16bf, u);
        }
#pragma unroll
        for (int j = 0; j < 2; ++j) {
            // B 32x16: lane n = l15; dwords 0..3 -> K = half*8 + 2v,
            //          dwords 4..7 -> K = 16 + half*8 + 2v
            int n = wn * 32 + j * 16 + l15;
            const uint32_t* brow = reinterpret_cast<const uint32_t*>(&Bs[n][0]);
            v8u u;
#pragma unroll
            for (int v = 0; v < 4; ++v) {
                u[v]     = brow[half * 4 + v];
                u[4 + v] = brow[8 + half * 4 + v];
            }
            bfrag[j] = __builtin_bit_cast(v16bf, u);
        }

#pragma unroll
        for (int i = 0; i < 2; ++i)
#pragma unroll
            for (int j = 0; j < 2; ++j)
                acc[i][j] = __builtin_amdgcn_wmma_f32_16x16x32_bf16(
                    false, afrag[i], false, bfrag[j],
                    (short)0, acc[i][j], false, false);
        __syncthreads();
    }

    // ---- store C (f32): VGPR e -> M = e + 8*half, N = l15 ----
#pragma unroll
    for (int i = 0; i < 2; ++i)
#pragma unroll
        for (int j = 0; j < 2; ++j) {
            int mb = m0 + wm * 32 + i * 16 + half * 8;
            int nb = n0 + wn * 32 + j * 16 + l15;
#pragma unroll
            for (int e = 0; e < 8; ++e)
                Cm[(size_t)(mb + e) * ldc + nb] = acc[i][j][e];
        }
}

// ---------------------------------------------------------------------------
// Column softmax over p of logit[p,q] = S[p,q]*rs[p]; out *= mm[p];
// writes bf16 transposed: T[q][p]  (A operand of GEMM2).
// One block = 64 consecutive columns, 256 threads (4 p-groups x 64 cols).
// ---------------------------------------------------------------------------
__global__ void __launch_bounds__(256)
k_softmax(const float* __restrict__ S, const float* __restrict__ rs,
          const float* __restrict__ mmv, uint16_t* __restrict__ T)
{
    __shared__ float red[4][64];
    __shared__ float colmax[64], colsum[64];
    const int qi = threadIdx.x & 63;
    const int pg = threadIdx.x >> 6;     // 0..3
    const int q  = blockIdx.x * 64 + qi;

    float m = -3.4e38f;
    for (int p = pg; p < Lp; p += 4)
        m = fmaxf(m, S[(size_t)p * Lp + q] * rs[p]);
    red[pg][qi] = m;
    __syncthreads();
    if (pg == 0)
        colmax[qi] = fmaxf(fmaxf(red[0][qi], red[1][qi]),
                           fmaxf(red[2][qi], red[3][qi]));
    __syncthreads();
    const float cm = colmax[qi];

    float s = 0.f;
    for (int p = pg; p < Lp; p += 4)
        s += __expf(S[(size_t)p * Lp + q] * rs[p] - cm);
    red[pg][qi] = s;
    __syncthreads();
    if (pg == 0)
        colsum[qi] = red[0][qi] + red[1][qi] + red[2][qi] + red[3][qi];
    __syncthreads();
    const float inv = 1.f / colsum[qi];

    for (int p = pg; p < Lp; p += 4) {
        float v = S[(size_t)p * Lp + q] * rs[p];
        float o = mmv[p] * __expf(v - cm) * inv;
        T[(size_t)q * Lp + p] = f2bf(o);
    }
}

// ---------------------------------------------------------------------------
// col2im: y[c,u,v] = 0.25 * sum over (a,b) with u=2i+a-1, v=2j+b-1 of
//                    Ycol[(i*64+j)][c*16 + a*4 + b]
// ---------------------------------------------------------------------------
__global__ void k_col2im(const float* __restrict__ Ycol, float* __restrict__ out, int b)
{
    int idx = blockIdx.x * blockDim.x + threadIdx.x;   // c*H*W + u*W + v
    int v = idx & 127;
    int u = (idx >> 7) & 127;
    int c = idx >> 14;
    float acc = 0.f;
#pragma unroll
    for (int a = 0; a < 4; ++a) {
        int iu = u + 1 - a;
        if (iu & 1) continue;
        int i = iu >> 1;
        if (i < 0 || i >= 64) continue;
#pragma unroll
        for (int bb = 0; bb < 4; ++bb) {
            int jv = v + 1 - bb;
            if (jv & 1) continue;
            int j = jv >> 1;
            if (j < 0 || j >= 64) continue;
            acc += Ycol[(size_t)(i * 64 + j) * K2D + c * 16 + a * 4 + bb];
        }
    }
    out[((size_t)b * Cc + c) * (size_t)(Hh * Ww) + (size_t)u * Ww + v] = acc * 0.25f;
}

// ---------------------------------------------------------------------------
// Host side: stream-ordered pipeline, workspace reused across batches.
// ---------------------------------------------------------------------------
extern "C" void kernel_launch(void* const* d_in, const int* in_sizes, int n_in,
                              void* d_out, int out_size, void* d_ws, size_t ws_size,
                              hipStream_t stream)
{
    (void)in_sizes; (void)n_in; (void)out_size; (void)ws_size;
    const float* x    = (const float*)d_in[0];   // [8,128,128,128]
    const float* mask = (const float*)d_in[1];   // [8,1,128,128]
    float*       out  = (float*)d_out;           // [8,128,128,128]

    char* ws = (char*)d_ws;
    // workspace layout (256B aligned), ~121 MB total, reused per batch:
    uint16_t* P    = (uint16_t*)(ws + 0);               //  9,437,184  P  [4096x1152] bf16
    uint16_t* W4   = (uint16_t*)(ws + 9437184);         // 16,777,216  W4 [4096x2048] bf16
    float*    rs   = (float*)   (ws + 26214400);        //     16,384
    float*    mmv  = (float*)   (ws + 26230784);        //     16,384
    float*    S    = (float*)   (ws + 26247168);        // 67,108,864  S  [4096x4096] f32
    uint16_t* T    = (uint16_t*)(ws + 93356032);        // 33,554,432  T  [4096x4096] bf16
    float*    Ycol = S;  // alias: S dead after softmax; Ycol needs 32 MB <= 64 MB

    for (int b = 0; b < 8; ++b) {
        k_prep3<<<Lp, 128, 0, stream>>>(x, mask, P, rs, mmv, b);
        k_prep4<<<Lp, 128, 0, stream>>>(x, W4, b);

        // GEMM1: S[p,q] = sum_k P[p,k] * P[q,k]   (M=N=4096, K=1152)
        dim3 g1(Lp / 128, Lp / 64);
        k_gemm_bf16<true><<<g1, 256, 0, stream>>>(P, K1D, P, K1D, S, Lp, K1D);

        // softmax over p per column q -> T[q][p] (bf16)
        k_softmax<<<Lp / 64, 256, 0, stream>>>(S, rs, mmv, T);

        // GEMM2: Ycol[q,n] = sum_p T[q,p] * W4[p,n]  (M=4096, N=2048, K=4096)
        dim3 g2(K2D / 128, Lp / 64);
        k_gemm_bf16<false><<<g2, 256, 0, stream>>>(T, Lp, W4, K2D, Ycol, K2D, Lp);

        // scatter back to full-res image, /4
        k_col2im<<<(Cc * Hh * Ww) / 256, 256, 0, stream>>>(Ycol, out, b);
    }
}
